// LSTMModel_31971736552251
// MI455X (gfx1250) — compile-verified, matrix operand order checked by reading, not run on hardware
//
#include <hip/hip_runtime.h>
#include <hip/hip_bf16.h>

// ---------------------------------------------------------------------------
// Types for WMMA fragments (wave32, V_WMMA_F32_16X16X32_F16)
// ---------------------------------------------------------------------------
typedef __attribute__((ext_vector_type(16))) _Float16 v16h;
typedef __attribute__((ext_vector_type(8)))  float    v8f;
typedef __attribute__((ext_vector_type(4)))  unsigned int u32x4;

union F16x16 { v16h v; u32x4 u[2]; };
union F16x8  { u32x4 u; _Float16 h[8]; };

#define T_SEQ   512
#define BATCH   256
#define HID     128
#define ROWS_WG 32                 // batch rows per workgroup
#define BLOCK   512                // 16 waves of 32

// Packed-weight region offsets (in halves)
#define R_L0F 0
#define R_L0B 81920                // 160*512
#define R_L1F 163840
#define R_L1B 360448               // +384*512
#define R_BIAS 557056
#define R_END  559104

// Branch-free fast transcendentals (keep the sequential scan loop short):
//   v_exp_f32 + v_rcp_f32, no EXEC-mask slow paths.
__device__ __forceinline__ float sigm(float x) {
  return __builtin_amdgcn_rcpf(1.0f + __expf(-x));
}
__device__ __forceinline__ float tanh_fast(float x) {
  return 1.0f - 2.0f * __builtin_amdgcn_rcpf(1.0f + __expf(2.0f * x));
}

// ---------------------------------------------------------------------------
// Tensor Data Mover: 2-D tile between LDS and global (fp16 elements).
// D# per CDNA5 ISA; groups built from uniform values
// so they materialize in SGPRs. Issued by wave 0 only (TENSORcnt per-wave).
// ---------------------------------------------------------------------------
#if __has_builtin(__builtin_amdgcn_tensor_load_to_lds) && \
    __has_builtin(__builtin_amdgcn_tensor_store_from_lds)
#define HAVE_TDM 1
#else
#define HAVE_TDM 0
#endif

#if HAVE_TDM
typedef __attribute__((ext_vector_type(4))) unsigned int sdw4u;
typedef __attribute__((ext_vector_type(8))) int          sdw8;
typedef __attribute__((ext_vector_type(4))) int          sdw4;

__device__ __forceinline__ void tdm_2d_f16(bool store, unsigned lds_off,
                                           unsigned long long gaddr,
                                           unsigned tile_w, unsigned tile_h,
                                           unsigned tensor_w, unsigned tensor_h,
                                           unsigned long long rstride_elems)
{
  sdw4u g0;
  g0.x = 1u;                                               // count=1 (valid D#)
  g0.y = lds_off;                                          // lds_addr (bytes)
  g0.z = (unsigned)(gaddr & 0xffffffffull);                // global_addr[31:0]
  g0.w = (unsigned)((gaddr >> 32) & 0x01ffffffull)         // global_addr[56:32]
         | (2u << 30);                                     // type=2 ("image")
  sdw8 g1;
  g1[0] = (int)(1u << 16);                                 // data_size=1 (2B), mask=0
  g1[1] = (int)((tensor_w & 0xffffu) << 16);               // tensor_dim0[15:0] @63:48
  g1[2] = (int)(((tensor_w >> 16) & 0xffffu)               // tensor_dim0[31:16]
                | ((tensor_h & 0xffffu) << 16));           // tensor_dim1[15:0]
  g1[3] = (int)(((tensor_h >> 16) & 0xffffu)               // tensor_dim1[31:16]
                | ((tile_w & 0xffffu) << 16));             // tile_dim0
  g1[4] = (int)(tile_h & 0xffffu);                         // tile_dim1 (tile_dim2=0)
  g1[5] = (int)(unsigned)(rstride_elems & 0xffffffffull);  // tensor_dim0_stride lo
  g1[6] = (int)(unsigned)((rstride_elems >> 32) & 0xffffull);
  g1[7] = 0;
  sdw4 z4 = 0;
#if defined(__clang_major__) && (__clang_major__ >= 23)
  sdw8 z8 = 0;
  if (store) __builtin_amdgcn_tensor_store_from_lds(g0, g1, z4, z4, z8, 0);
  else       __builtin_amdgcn_tensor_load_to_lds  (g0, g1, z4, z4, z8, 0);
#else
  if (store) __builtin_amdgcn_tensor_store_from_lds(g0, g1, z4, z4, 0);
  else       __builtin_amdgcn_tensor_load_to_lds  (g0, g1, z4, z4, 0);
#endif
}

__device__ __forceinline__ void tdm_wait0() {
#if __has_builtin(__builtin_amdgcn_s_wait_tensorcnt)
  __builtin_amdgcn_s_wait_tensorcnt(0);
#else
  asm volatile("s_wait_tensorcnt 0x0" ::: "memory");
#endif
}
#endif // HAVE_TDM

// ---------------------------------------------------------------------------
// Pack Whh/Wih into per-lane WMMA B-fragment layout (fp16) + combined biases.
// Fragment (kt,nt): 32 lanes x 16 halves contiguous. For lane L, half h:
//   k = kt*32 + (h&7) + ((L>>4)<<3) + ((h>>3)<<4),  n = nt*16 + (L&15)
// ---------------------------------------------------------------------------
__global__ void pack_kernel(
    const float* l0f_wih, const float* l0f_whh, const float* l0f_bih, const float* l0f_bhh,
    const float* l0b_wih, const float* l0b_whh, const float* l0b_bih, const float* l0b_bhh,
    const float* l1f_wih, const float* l1f_whh, const float* l1f_bih, const float* l1f_bhh,
    const float* l1b_wih, const float* l1b_whh, const float* l1b_bih, const float* l1b_bhh,
    _Float16* __restrict__ wpack, float* __restrict__ biasc)
{
  int idx = blockIdx.x * blockDim.x + threadIdx.x;
  if (idx >= R_END) return;

  if (idx >= R_BIAS) {                       // combined bias = bih + bhh
    int i2 = idx - R_BIAS;
    int set = i2 >> 9, n = i2 & 511;
    const float *bih, *bhh;
    if      (set == 0) { bih = l0f_bih; bhh = l0f_bhh; }
    else if (set == 1) { bih = l0b_bih; bhh = l0b_bhh; }
    else if (set == 2) { bih = l1f_bih; bhh = l1f_bhh; }
    else               { bih = l1b_bih; bhh = l1b_bhh; }
    biasc[set * 512 + n] = bih[n] + bhh[n];
    return;
  }

  const float *whh, *wih; int kin, local;
  if      (idx < R_L0B) { whh = l0f_whh; wih = l0f_wih; kin = 6;   local = idx - R_L0F; }
  else if (idx < R_L1F) { whh = l0b_whh; wih = l0b_wih; kin = 6;   local = idx - R_L0B; }
  else if (idx < R_L1B) { whh = l1f_whh; wih = l1f_wih; kin = 256; local = idx - R_L1F; }
  else                  { whh = l1b_whh; wih = l1b_wih; kin = 256; local = idx - R_L1B; }

  int frag = local >> 9, q = local & 511;
  int lane = q >> 4,  h = q & 15;
  int kt = frag >> 5, nt = frag & 31;
  int k = kt * 32 + (h & 7) + ((lane >> 4) << 3) + ((h >> 3) << 4);
  int n = nt * 16 + (lane & 15);
  float v = 0.0f;
  if (k < 128)            v = whh[n * 128 + k];            // Whh^T (K rows 0..127)
  else if (k < 128 + kin) v = wih[n * kin + (k - 128)];    // Wih^T
  wpack[idx] = (_Float16)v;
}

// ---------------------------------------------------------------------------
// Layer-0 bidirectional LSTM scan. grid.x = 16 (8 batch slices x 2 dirs).
// K = 160 padded (h:0..127, x:128..133, zeros..159) -> 5 WMMA k-steps.
// h tile writeback via TDM store from the LDS double buffer.
// ---------------------------------------------------------------------------
__global__ void __launch_bounds__(BLOCK)
lstm_l0_kernel(const float* __restrict__ x, const _Float16* __restrict__ wpack,
               const float* __restrict__ biasc, _Float16* __restrict__ h0)
{
  extern __shared__ _Float16 smem0[];
  _Float16* hbuf = smem0;                     // [2][32][128]  LDS off 0 / 8192
  _Float16* xbuf = smem0 + 2 * ROWS_WG * HID; // [32][32]      LDS off 16384

  const int tid   = threadIdx.x;
  const int dir   = blockIdx.x & 1;
  const int slice = blockIdx.x >> 1;
  const _Float16* wp = wpack + (dir ? R_L0B : R_L0F);
  const float*    bp = biasc + dir * 512;
  const unsigned long long h0base =
      (unsigned long long)h0 + (unsigned long long)(2 * HID * dir); // +dir*128 elems

  for (int i = tid; i < 2 * ROWS_WG * HID + ROWS_WG * 32; i += BLOCK)
    smem0[i] = (_Float16)0.0f;
  __syncthreads();

  const int lane = tid & 31, wave = tid >> 5;
  const int mt = wave >> 3, ntg = wave & 7;
  const int lm = lane & 15, lh = lane >> 4;
  const int arow = mt * 16 + lm;

  float cst[8];
  #pragma unroll
  for (int e = 0; e < 8; ++e) cst[e] = 0.0f;

  float bgate[4];
  #pragma unroll
  for (int g = 0; g < 4; ++g) bgate[g] = bp[(g * 8 + ntg) * 16 + lm];

  int buf = 0;
  for (int tt = 0; tt < T_SEQ; ++tt) {
    const int t = dir ? (T_SEQ - 1 - tt) : tt;

    for (int i = tid; i < ROWS_WG * 6; i += BLOCK) {   // stage x_t (Din=6, f32->f16)
      int r = i / 6, c = i - r * 6;
      xbuf[r * 32 + c] =
          (_Float16)x[((size_t)(slice * ROWS_WG + r) * T_SEQ + t) * 6 + c];
    }
#if HAVE_TDM
    if (wave == 0) tdm_wait0();   // h-store from two steps ago must be done
#endif
    __syncthreads();

    v8f acc[4];
    #pragma unroll
    for (int g = 0; g < 4; ++g) {
      #pragma unroll
      for (int e = 0; e < 8; ++e) acc[g][e] = bgate[g];
    }

    #pragma unroll
    for (int kt = 0; kt < 5; ++kt) {
      F16x16 af;
      const _Float16* ap = (kt < 4)
          ? &hbuf[(buf * ROWS_WG + arow) * HID + kt * 32 + lh * 8]
          : &xbuf[arow * 32 + lh * 8];
      af.u[0] = *reinterpret_cast<const u32x4*>(ap);
      af.u[1] = *reinterpret_cast<const u32x4*>(ap + 16);
      #pragma unroll
      for (int g = 0; g < 4; ++g) {
        F16x16 bf;
        const _Float16* bq = wp + (((size_t)(kt * 32 + (g * 8 + ntg)) * 32 + lane) << 4);
        bf.u[0] = *reinterpret_cast<const u32x4*>(bq);
        bf.u[1] = *reinterpret_cast<const u32x4*>(bq + 8);
        acc[g] = __builtin_amdgcn_wmma_f32_16x16x32_f16(
            false, af.v, false, bf.v, (short)0, acc[g], false, false);
      }
    }

    const int nbuf = buf ^ 1;
    #pragma unroll
    for (int e = 0; e < 8; ++e) {
      float iv = sigm(acc[0][e]);
      float fv = sigm(acc[1][e]);
      float gv = tanh_fast(acc[2][e]);
      float ov = sigm(acc[3][e]);
      float c  = fv * cst[e] + iv * gv;
      cst[e] = c;
      float h  = ov * tanh_fast(c);
      int r = mt * 16 + lh * 8 + e;
      int j = ntg * 16 + lm;
      _Float16 hh = (_Float16)h;
      hbuf[(nbuf * ROWS_WG + r) * HID + j] = hh;
#if !HAVE_TDM
      h0[((size_t)(slice * ROWS_WG + r) * T_SEQ + t) * 256 + dir * HID + j] = hh;
#endif
    }
    __syncthreads();
#if HAVE_TDM
    if (wave == 0) {               // DMA h tile [32][128] -> h0[:, t, dir*128:+128]
      unsigned long long g =
          h0base + ((unsigned long long)((size_t)(slice * ROWS_WG) * T_SEQ + t) * 256ull) * 2ull;
      tdm_2d_f16(true, (unsigned)(nbuf * ROWS_WG * HID * 2), g,
                 HID, ROWS_WG, HID, ROWS_WG, (unsigned long long)T_SEQ * 256ull);
    }
#endif
    buf = nbuf;
  }
}

// ---------------------------------------------------------------------------
// Layer-1 bidirectional LSTM. Input projection fused: K = 128 (h) + 256 (x)
// -> 12 WMMA k-steps. x_t = h0[:,t,:] staged to LDS by TDM (double-buffered,
// prefetching t+1 during compute of t); h writeback via TDM store.
// ---------------------------------------------------------------------------
__global__ void __launch_bounds__(BLOCK)
lstm_l1_kernel(const _Float16* __restrict__ h0, const _Float16* __restrict__ wpack,
               const float* __restrict__ biasc, _Float16* __restrict__ h1)
{
  extern __shared__ _Float16 smem1[];
  _Float16* hbuf  = smem1;                      // [2][32][128]  LDS off 0 / 8192
  _Float16* xtbuf = smem1 + 2 * ROWS_WG * HID;  // [2][32][256]  LDS off 16384 / 32768

  const int tid   = threadIdx.x;
  const int dir   = blockIdx.x & 1;
  const int slice = blockIdx.x >> 1;
  const _Float16* wp = wpack + (dir ? R_L1B : R_L1F);
  const float*    bp = biasc + (2 + dir) * 512;
  const unsigned long long h0base = (unsigned long long)h0;
  const unsigned long long h1base =
      (unsigned long long)h1 + (unsigned long long)(2 * HID * dir);

  for (int i = tid; i < 2 * ROWS_WG * HID; i += BLOCK) hbuf[i] = (_Float16)0.0f;
  __syncthreads();

  const int lane = tid & 31, wave = tid >> 5;
  const int mt = wave >> 3, ntg = wave & 7;
  const int lm = lane & 15, lh = lane >> 4;
  const int arow = mt * 16 + lm;

  float cst[8];
  #pragma unroll
  for (int e = 0; e < 8; ++e) cst[e] = 0.0f;

  float bgate[4];
  #pragma unroll
  for (int g = 0; g < 4; ++g) bgate[g] = bp[(g * 8 + ntg) * 16 + lm];

  int buf = 0, xcur = 0;
#if HAVE_TDM
  if (wave == 0) {                 // kick off x_t DMA for the first timestep
    const int t0 = dir ? (T_SEQ - 1) : 0;
    unsigned long long g =
        h0base + ((unsigned long long)((size_t)(slice * ROWS_WG) * T_SEQ + t0) * 256ull) * 2ull;
    tdm_2d_f16(false, (unsigned)((2 * ROWS_WG * HID) * 2), g,
               256, ROWS_WG, 256, ROWS_WG, (unsigned long long)T_SEQ * 256ull);
  }
#endif

  for (int tt = 0; tt < T_SEQ; ++tt) {
    const int t = dir ? (T_SEQ - 1 - tt) : tt;

#if HAVE_TDM
    if (wave == 0) tdm_wait0();    // x_t in LDS + older h stores retired
    __syncthreads();
    if (wave == 0 && tt + 1 < T_SEQ) {  // prefetch x_{t+1} into the other buffer
      const int tn = dir ? (T_SEQ - 2 - tt) : (tt + 1);
      unsigned long long g =
          h0base + ((unsigned long long)((size_t)(slice * ROWS_WG) * T_SEQ + tn) * 256ull) * 2ull;
      tdm_2d_f16(false,
                 (unsigned)((2 * ROWS_WG * HID + (xcur ^ 1) * ROWS_WG * 256) * 2), g,
                 256, ROWS_WG, 256, ROWS_WG, (unsigned long long)T_SEQ * 256ull);
    }
#else
    for (int i = tid; i < ROWS_WG * 32; i += BLOCK) {
      int r = i >> 5, c = i & 31;
      reinterpret_cast<u32x4*>(xtbuf + xcur * ROWS_WG * 256 + r * 256)[c] =
          reinterpret_cast<const u32x4*>(
              h0 + ((size_t)(slice * ROWS_WG + r) * T_SEQ + t) * 256)[c];
    }
    __syncthreads();
#endif
    const _Float16* xt = xtbuf + xcur * ROWS_WG * 256;

    v8f acc[4];
    #pragma unroll
    for (int g = 0; g < 4; ++g) {
      #pragma unroll
      for (int e = 0; e < 8; ++e) acc[g][e] = bgate[g];
    }

    #pragma unroll
    for (int kt = 0; kt < 12; ++kt) {
      F16x16 af;
      const _Float16* ap = (kt < 4)
          ? &hbuf[(buf * ROWS_WG + arow) * HID + kt * 32 + lh * 8]
          : &xt[arow * 256 + (kt - 4) * 32 + lh * 8];
      af.u[0] = *reinterpret_cast<const u32x4*>(ap);
      af.u[1] = *reinterpret_cast<const u32x4*>(ap + 16);
      #pragma unroll
      for (int g = 0; g < 4; ++g) {
        F16x16 bf;
        const _Float16* bq = wp + (((size_t)(kt * 32 + (g * 8 + ntg)) * 32 + lane) << 4);
        bf.u[0] = *reinterpret_cast<const u32x4*>(bq);
        bf.u[1] = *reinterpret_cast<const u32x4*>(bq + 8);
        acc[g] = __builtin_amdgcn_wmma_f32_16x16x32_f16(
            false, af.v, false, bf.v, (short)0, acc[g], false, false);
      }
    }

    const int nbuf = buf ^ 1;
    #pragma unroll
    for (int e = 0; e < 8; ++e) {
      float iv = sigm(acc[0][e]);
      float fv = sigm(acc[1][e]);
      float gv = tanh_fast(acc[2][e]);
      float ov = sigm(acc[3][e]);
      float c  = fv * cst[e] + iv * gv;
      cst[e] = c;
      float h  = ov * tanh_fast(c);
      int r = mt * 16 + lh * 8 + e;
      int j = ntg * 16 + lm;
      _Float16 hh = (_Float16)h;
      hbuf[(nbuf * ROWS_WG + r) * HID + j] = hh;
#if !HAVE_TDM
      h1[((size_t)(slice * ROWS_WG + r) * T_SEQ + t) * 256 + dir * HID + j] = hh;
#endif
    }
    __syncthreads();
#if HAVE_TDM
    if (wave == 0) {               // DMA h tile -> h1[:, t, dir*128:+128]
      unsigned long long g =
          h1base + ((unsigned long long)((size_t)(slice * ROWS_WG) * T_SEQ + t) * 256ull) * 2ull;
      tdm_2d_f16(true, (unsigned)(nbuf * ROWS_WG * HID * 2), g,
                 HID, ROWS_WG, HID, ROWS_WG, (unsigned long long)T_SEQ * 256ull);
    }
#endif
    buf = nbuf;
    xcur ^= 1;
  }
}

// ---------------------------------------------------------------------------
// Head: collapse (256->64->8->1) linears into u[256], bias; then y = h1.u + b
// ---------------------------------------------------------------------------
__global__ void head_prep_kernel(const float* __restrict__ lin1w, const float* __restrict__ lin1b,
                                 const float* __restrict__ lin2w, const float* __restrict__ lin2b,
                                 const float* __restrict__ outw,  const float* __restrict__ outb,
                                 float* __restrict__ u, float* __restrict__ hbias)
{
  __shared__ float v[64];
  int tid = threadIdx.x;
  if (tid < 64) {
    float s = 0.0f;
    for (int j = 0; j < 8; ++j) s += outw[j] * lin2w[j * 64 + tid];
    v[tid] = s;
  }
  __syncthreads();
  float s = 0.0f;
  for (int k = 0; k < 64; ++k) s += v[k] * lin1w[k * 256 + tid];
  u[tid] = s;
  if (tid == 0) {
    float b = outb[0];
    for (int k = 0; k < 64; ++k) b += lin1b[k] * v[k];
    for (int j = 0; j < 8; ++j)  b += lin2b[j] * outw[j];
    hbias[0] = b;
  }
}

__global__ void head_apply_kernel(const _Float16* __restrict__ h1,
                                  const float* __restrict__ u,
                                  const float* __restrict__ hbias,
                                  float* __restrict__ y)
{
  __shared__ float us[256];
  int tid = threadIdx.x;
  us[tid] = u[tid];
  __syncthreads();
  int bt = blockIdx.x * 256 + tid;
  const u32x4* hp = reinterpret_cast<const u32x4*>(h1 + (size_t)bt * 256);
  float acc = 0.0f;
  #pragma unroll 4
  for (int n8 = 0; n8 < 32; ++n8) {
    F16x8 pk; pk.u = hp[n8];
    #pragma unroll
    for (int e = 0; e < 8; ++e) acc += (float)pk.h[e] * us[n8 * 8 + e];
  }
  y[bt] = acc + hbias[0];
}

// ---------------------------------------------------------------------------
// Workspace layout (bytes):
//  0         wpack   557056 halves (1114112 B)
//  1114112   biasc   2048 f32      (8192 B)
//  1122304   h0      256*512*256 f16 (67108864 B)
//  68231168  h1      256*512*256 f16 (67108864 B)
//  135340032 u[256] + hbias[1] f32
// ---------------------------------------------------------------------------
extern "C" void kernel_launch(void* const* d_in, const int* in_sizes, int n_in,
                              void* d_out, int out_size, void* d_ws, size_t ws_size,
                              hipStream_t stream) {
  (void)in_sizes; (void)n_in; (void)out_size; (void)ws_size;
  const float* x = (const float*)d_in[0];
  const float* p[22];
  for (int i = 0; i < 22; ++i) p[i] = (const float*)d_in[i + 1];

  char* ws = (char*)d_ws;
  _Float16* wpack = (_Float16*)ws;
  float*    biasc = (float*)(ws + 1114112);
  _Float16* h0    = (_Float16*)(ws + 1122304);
  _Float16* h1    = (_Float16*)(ws + 68231168);
  float*    uvec  = (float*)(ws + 135340032);
  float*    hbias = uvec + 256;

  pack_kernel<<<(R_END + 255) / 256, 256, 0, stream>>>(
      p[0], p[1], p[2], p[3],   p[4], p[5], p[6], p[7],
      p[8], p[9], p[10], p[11], p[12], p[13], p[14], p[15],
      wpack, biasc);

  const size_t smem0 = (size_t)(2 * ROWS_WG * HID + ROWS_WG * 32) * sizeof(_Float16);
  const size_t smem1 = (size_t)(2 * ROWS_WG * HID + 2 * ROWS_WG * 256) * sizeof(_Float16);
  lstm_l0_kernel<<<16, BLOCK, smem0, stream>>>(x, wpack, biasc, h0);
  lstm_l1_kernel<<<16, BLOCK, smem1, stream>>>(h0, wpack, biasc, h1);

  head_prep_kernel<<<1, 256, 0, stream>>>(p[16], p[17], p[18], p[19], p[20], p[21],
                                          uvec, hbias);
  head_apply_kernel<<<512, 256, 0, stream>>>(h1, uvec, hbias, (float*)d_out);
}
